// transformer_45406394254086
// MI455X (gfx1250) — matrix operand order repaired and checked by reading, not compiled, not run on hardware
//
#include <hip/hip_runtime.h>
#include <math.h>

// ---------------------------------------------------------------------------
// CDNA5 (gfx1250) WMMA transformer block. wave32 only.
// ---------------------------------------------------------------------------

typedef __attribute__((ext_vector_type(16))) _Float16 v16h;
typedef __attribute__((ext_vector_type(8)))  _Float16 v8h;
typedef __attribute__((ext_vector_type(8)))  float    v8f;

static __device__ __forceinline__ v16h cat8(v8h a, v8h b) {
  return __builtin_shufflevector(a, b, 0,1,2,3,4,5,6,7,8,9,10,11,12,13,14,15);
}

static __device__ __forceinline__ v8f wmma16(v16h a, v16h b, v8f c) {
  // D = A(16x32 f16) x B(32x16 f16) + C(16x16 f32)
  return __builtin_amdgcn_wmma_f32_16x16x32_f16(
      /*neg_a=*/false, a, /*neg_b=*/false, b,
      /*c_mod=*/(short)0, c, /*reuse_a=*/false, /*reuse_b=*/false);
}

// ---------------------------------------------------------------------------
// WMMA GEMM: C[M,N](f32) = A[M,K](f16 row-major) * Bt[N,K](f16)^T
// Block tile 128x256x32, 8 waves, wave tile 64x64 (16 WMMAs / K-step / wave).
// LDS double-buffered via register staging: one barrier per K-step.
// EPI bit0: add bias[col]; bit1: add res[row,col].
// Requires M%128==0, N%256==0, K%32==0.
// ---------------------------------------------------------------------------
template <int EPI>
__global__ __launch_bounds__(256) void gemm_wmma(
    const _Float16* __restrict__ A, const _Float16* __restrict__ Bt,
    float* __restrict__ C, const float* __restrict__ bias,
    const float* __restrict__ res, int M, int N, int K) {
  constexpr int BM = 128, BN = 256, BK = 32;
  constexpr int LDT = 40;  // 32 halves + 8 pad (80B rows, 16B-aligned)
  __shared__ alignas(16) _Float16 sA[2][BM * LDT];
  __shared__ alignas(16) _Float16 sB[2][BN * LDT];

  const int tid  = threadIdx.x;
  const int wave = tid >> 5, lane = tid & 31;
  const int wr = wave >> 2, wc = wave & 3;   // 2x4 wave grid -> 64x64 wave tile
  const int lm = lane & 15, lh = lane >> 4;
  const int m0 = blockIdx.y * BM, n0 = blockIdx.x * BN;

  const int lrow = tid >> 1;          // 0..127
  const int lcol = (tid & 1) * 8;     // half offset (16B chunk)

  v8f acc[4][4];
#pragma unroll
  for (int i = 0; i < 4; ++i)
#pragma unroll
    for (int j = 0; j < 4; ++j) acc[i][j] = (v8f){};

  const _Float16* gA  = A  + (size_t)(m0 + lrow) * K + lcol;
  const _Float16* gB0 = Bt + (size_t)(n0 + lrow) * K + lcol;
  const _Float16* gB1 = Bt + (size_t)(n0 + 128 + lrow) * K + lcol;

  // prologue: tile 0 -> buffer 0
  *(uint4*)(sA[0] + lrow * LDT + lcol)         = *(const uint4*)(gA);
  *(uint4*)(sB[0] + lrow * LDT + lcol)         = *(const uint4*)(gB0);
  *(uint4*)(sB[0] + (128 + lrow) * LDT + lcol) = *(const uint4*)(gB1);
  __syncthreads();

  const int nk = K / BK;
  for (int kt = 0; kt < nk; ++kt) {
    const int cur = kt & 1, nxt = cur ^ 1;
    uint4 ra = {}, rb0 = {}, rb1 = {};
    if (kt + 1 < nk) {  // issue next-tile global loads before the WMMAs
      const int k1 = (kt + 1) * BK;
      ra  = *(const uint4*)(gA  + k1);
      rb0 = *(const uint4*)(gB0 + k1);
      rb1 = *(const uint4*)(gB1 + k1);
    }

    v16h af[4], bf[4];
#pragma unroll
    for (int rt = 0; rt < 4; ++rt) {
      // A frag 16x32: lane(m=lm,h=lh); v0..3 <- K=lh*8..+7, v4..7 <- K=16+lh*8..+7
      const _Float16* p = sA[cur] + (wr * 64 + rt * 16 + lm) * LDT + lh * 8;
      af[rt] = cat8(*(const v8h*)p, *(const v8h*)(p + 16));
    }
#pragma unroll
    for (int ct = 0; ct < 4; ++ct) {
      // B frag 32x16: lane(n=lm,h=lh); v0..3 <- K=lh*8..+7, v4..7 <- K=16+lh*8..+7
      const _Float16* p = sB[cur] + (wc * 64 + ct * 16 + lm) * LDT + lh * 8;
      bf[ct] = cat8(*(const v8h*)p, *(const v8h*)(p + 16));
    }
#pragma unroll
    for (int rt = 0; rt < 4; ++rt)
#pragma unroll
      for (int ct = 0; ct < 4; ++ct)
        acc[rt][ct] = wmma16(af[rt], bf[ct], acc[rt][ct]);

    if (kt + 1 < nk) {  // stores to the *other* buffer: no barrier needed first
      *(uint4*)(sA[nxt] + lrow * LDT + lcol)         = ra;
      *(uint4*)(sB[nxt] + lrow * LDT + lcol)         = rb0;
      *(uint4*)(sB[nxt] + (128 + lrow) * LDT + lcol) = rb1;
    }
    __syncthreads();
  }

#pragma unroll
  for (int rt = 0; rt < 4; ++rt)
#pragma unroll
    for (int ct = 0; ct < 4; ++ct) {
      const int col = n0 + wc * 64 + ct * 16 + lm;
      float bv = (EPI & 1) ? bias[col] : 0.0f;
#pragma unroll
      for (int r = 0; r < 8; ++r) {
        const int row = m0 + wr * 64 + rt * 16 + r + lh * 8;
        float v = acc[rt][ct][r] + bv;
        if (EPI & 2) v += res[(size_t)row * N + col];
        C[(size_t)row * N + col] = v;
      }
    }
}

// ---------------------------------------------------------------------------
// LayerNorm: per row of 1024, fp32 in -> f16 out (GEMM A operand)
// ---------------------------------------------------------------------------
__global__ __launch_bounds__(256) void ln_kernel(
    const float* __restrict__ x, const float* __restrict__ g,
    const float* __restrict__ b, _Float16* __restrict__ out) {
  constexpr int D = 1024;
  __shared__ float red[256];
  const int row = blockIdx.x, t = threadIdx.x;
  const float* xr = x + (size_t)row * D;
  float v[4], s = 0.f;
#pragma unroll
  for (int i = 0; i < 4; ++i) { v[i] = xr[t + i * 256]; s += v[i]; }
  red[t] = s; __syncthreads();
  for (int o = 128; o > 0; o >>= 1) { if (t < o) red[t] += red[t + o]; __syncthreads(); }
  const float mean = red[0] * (1.0f / D);
  __syncthreads();
  float q = 0.f;
#pragma unroll
  for (int i = 0; i < 4; ++i) { float d = v[i] - mean; q += d * d; }
  red[t] = q; __syncthreads();
  for (int o = 128; o > 0; o >>= 1) { if (t < o) red[t] += red[t + o]; __syncthreads(); }
  const float rstd = rsqrtf(red[0] * (1.0f / D) + 1e-5f);
#pragma unroll
  for (int i = 0; i < 4; ++i) {
    const int c = t + i * 256;
    out[(size_t)row * D + c] = (_Float16)((v[i] - mean) * rstd * g[c] + b[c]);
  }
}

// ---------------------------------------------------------------------------
// fp32 [K,N] weight -> f16 [N,K]  (transposed for K-contiguous LDS tiles)
// ---------------------------------------------------------------------------
__global__ __launch_bounds__(256) void wconv_kernel(
    const float* __restrict__ w, _Float16* __restrict__ wt, int K, int N) {
  const size_t idx = (size_t)blockIdx.x * 256 + threadIdx.x;
  const size_t k = idx / (size_t)N, n = idx % (size_t)N;
  wt[n * (size_t)K + k] = (_Float16)w[idx];
}

// qkv weight [K, (hd*3+s)] -> f16 [c'=s*1024+hd][K]; GEMM then emits
// plane-major qkv [q|k|v] so attention loads are 128-bit coalesced.
__global__ __launch_bounds__(256) void wconv_qkv_kernel(
    const float* __restrict__ w, _Float16* __restrict__ wt, int K) {
  const size_t idx = (size_t)blockIdx.x * 256 + threadIdx.x;
  const size_t k = idx / 3072, c = idx % 3072;
  const size_t hd = c / 3, s = c % 3;
  wt[(s * 1024 + hd) * (size_t)K + k] = (_Float16)w[idx];
}

__global__ __launch_bounds__(256) void copy_kernel(
    const float* __restrict__ a, float* __restrict__ b) {
  const size_t i = (size_t)blockIdx.x * 256 + threadIdx.x;
  b[i] = a[i];
}

// a * silu(g), p[M,8192] -> out[M,4096] f16
__global__ __launch_bounds__(256) void glu_kernel(
    const float* __restrict__ p, _Float16* __restrict__ out) {
  const size_t idx = (size_t)blockIdx.x * 256 + threadIdx.x;
  const size_t row = idx >> 12, col = idx & 4095;
  const float a = p[row * 8192 + col];
  const float g = p[row * 8192 + 4096 + col];
  out[idx] = (_Float16)(a * (g / (1.0f + __expf(-g))));
}

// ---------------------------------------------------------------------------
// Dynamic position bias MLP -> tab[H][2N-1]
// ---------------------------------------------------------------------------
__global__ __launch_bounds__(256) void posbias_kernel(
    const float* __restrict__ p1w, const float* __restrict__ p1b,
    const float* __restrict__ p2w, const float* __restrict__ p2b,
    const float* __restrict__ p3w, const float* __restrict__ p3b,
    float* __restrict__ tab, int n) {
  __shared__ float h1[256];
  __shared__ float h2[256];
  __shared__ float red[256];
  const int rowi = blockIdx.x;              // 0 .. 2n-2
  const float pos = (float)(rowi - (n - 1));
  const int t = threadIdx.x;
  float a = pos * p1w[t] + p1b[t];
  h1[t] = a > 0.f ? a : 0.f;
  __syncthreads();
  float acc = p2b[t];
  for (int i = 0; i < 256; ++i) acc += h1[i] * p2w[i * 256 + t];
  h2[t] = acc > 0.f ? acc : 0.f;
  __syncthreads();
  for (int h = 0; h < 16; ++h) {
    red[t] = h2[t] * p3w[t * 16 + h];
    __syncthreads();
    for (int o = 128; o > 0; o >>= 1) { if (t < o) red[t] += red[t + o]; __syncthreads(); }
    if (t == 0) tab[(size_t)h * (2 * n - 1) + rowi] = red[0] + p3b[h];
    __syncthreads();
  }
}

// ---------------------------------------------------------------------------
// Flash attention (cosine attn): one WG per (b, head, 128-row q tile).
// 8 waves x 16 q-rows, temp folded into l2-normalized q, online softmax.
// qkv is PLANE-MAJOR: [B*N, 3*1024], plane s at s*1024, col hh*64+d.
// ---------------------------------------------------------------------------
__global__ __launch_bounds__(256) void attn_kernel(
    const float* __restrict__ qkv, const float* __restrict__ tab,
    const unsigned char* __restrict__ mask, const float* __restrict__ tptr,
    _Float16* __restrict__ o16) {
  constexpr int N = 1024;
  __shared__ alignas(16) _Float16 sK[32 * 64];   // [key][d], l2-normalized
  __shared__ alignas(16) _Float16 sV[64 * 32];   // [d][key]  (transposed)
  __shared__ alignas(16) _Float16 sP[8][16 * 32];

  const int wg = blockIdx.x;
  const int qt = wg & 7, hh = (wg >> 3) & 15, b = wg >> 7;
  const int tid = threadIdx.x, wave = tid >> 5, lane = tid & 31;
  const int lm = lane & 15, lh = lane >> 4;
  const float temp = tptr[0];

  // ---- Q fragments (A layout 16x32 per d-chunk), normalize + fold temp ----
  const int qrow = qt * 128 + wave * 16 + lm;
  const float* qb = qkv + (size_t)(b * N + qrow) * 3072 + hh * 64;  // q plane
  float qv[2][16];
  float ss = 0.f;
#pragma unroll
  for (int c = 0; c < 2; ++c)
#pragma unroll
    for (int v = 0; v < 8; ++v) {
      const int d = c * 32 + (v >> 2) * 16 + lh * 8 + (v & 3) * 2;
      const float2 x01 = *(const float2*)(qb + d);
      qv[c][2 * v] = x01.x; qv[c][2 * v + 1] = x01.y;
      ss += x01.x * x01.x + x01.y * x01.y;
    }
  ss += __shfl_xor(ss, 16, 32);
  const float qs = temp / fmaxf(sqrtf(ss), 1e-12f);
  v16h qf[2];
#pragma unroll
  for (int c = 0; c < 2; ++c) {
    v16h f;
#pragma unroll
    for (int i = 0; i < 16; ++i) f[i] = (_Float16)(qv[c][i] * qs);
    qf[c] = f;
  }

  // per-lane rows in the C-fragment layout: r + 8*lh
  int  qg_r[8];
  bool vq_r[8];
#pragma unroll
  for (int r = 0; r < 8; ++r) {
    qg_r[r] = qt * 128 + wave * 16 + r + 8 * lh;
    vq_r[r] = (mask[b * N + qg_r[r]] == 0);
  }

  float m_r[8], s_r[8];
  v8f Oacc[4];
#pragma unroll
  for (int r = 0; r < 8; ++r) { m_r[r] = -3.0e38f; s_r[r] = 0.f; }
#pragma unroll
  for (int g2 = 0; g2 < 4; ++g2) Oacc[g2] = (v8f){};

  // cooperative-load addressing: 8 threads per key row, 8 d's each
  const int crow = tid >> 3, cj = tid & 7;
  const float* kbase = qkv + (size_t)(b * N + crow) * 3072 + 1024 + hh * 64 + cj * 8;
  const float* vbase = kbase + 1024;
  constexpr size_t ROWSTEP = (size_t)32 * 3072;  // 32 key rows

  const int nkb = (qt + 1) * 4;  // causal: keys up to end of this q-tile
  for (int kb = 0; kb < nkb; ++kb) {
    const int k0 = kb * 32;
    __syncthreads();
    {  // K/V tile load: two float4 per thread per tensor, fully coalesced
      const float4 ka = *(const float4*)(kbase + (size_t)kb * ROWSTEP);
      const float4 kc = *(const float4*)(kbase + (size_t)kb * ROWSTEP + 4);
      const float4 va = *(const float4*)(vbase + (size_t)kb * ROWSTEP);
      const float4 vc = *(const float4*)(vbase + (size_t)kb * ROWSTEP + 4);
      if (kb + 1 < nkb) {  // global_prefetch_b8 for next tile
        __builtin_prefetch(kbase + (size_t)(kb + 1) * ROWSTEP, 0, 1);
        __builtin_prefetch(vbase + (size_t)(kb + 1) * ROWSTEP, 0, 1);
      }
      float ks = ka.x * ka.x + ka.y * ka.y + ka.z * ka.z + ka.w * ka.w +
                 kc.x * kc.x + kc.y * kc.y + kc.z * kc.z + kc.w * kc.w;
      ks += __shfl_xor(ks, 1, 32);
      ks += __shfl_xor(ks, 2, 32);
      ks += __shfl_xor(ks, 4, 32);
      const float rn = 1.0f / fmaxf(sqrtf(ks), 1e-12f);
      _Float16* kd = sK + crow * 64 + cj * 8;
      kd[0] = (_Float16)(ka.x * rn); kd[1] = (_Float16)(ka.y * rn);
      kd[2] = (_Float16)(ka.z * rn); kd[3] = (_Float16)(ka.w * rn);
      kd[4] = (_Float16)(kc.x * rn); kd[5] = (_Float16)(kc.y * rn);
      kd[6] = (_Float16)(kc.z * rn); kd[7] = (_Float16)(kc.w * rn);
      _Float16* vd = sV + (cj * 8) * 32 + crow;      // transposed [d][key]
      vd[0 * 32] = (_Float16)va.x; vd[1 * 32] = (_Float16)va.y;
      vd[2 * 32] = (_Float16)va.z; vd[3 * 32] = (_Float16)va.w;
      vd[4 * 32] = (_Float16)vc.x; vd[5 * 32] = (_Float16)vc.y;
      vd[6 * 32] = (_Float16)vc.z; vd[7 * 32] = (_Float16)vc.w;
    }
    __syncthreads();

    // ---- scores: S(16q x 16k) = Q . K^T over d=64 (2 WMMAs per key tile)
    v8f sc[2];
#pragma unroll
    for (int tt = 0; tt < 2; ++tt) {
      const _Float16* p0 = sK + (tt * 16 + lm) * 64 + lh * 8;  // d-chunk 0
      const _Float16* p1 = p0 + 32;                             // d-chunk 1
      v16h k0f = cat8(*(const v8h*)p0, *(const v8h*)(p0 + 16));
      v16h k1f = cat8(*(const v8h*)p1, *(const v8h*)(p1 + 16));
      v8f z = (v8f){};
      sc[tt] = wmma16(qf[0], k0f, wmma16(qf[1], k1f, z));
    }

    // ---- bias + mask + online softmax (stats uniform per 16-lane half) ----
    const int kg0 = k0 + lm, kg1 = k0 + 16 + lm;
    const bool vk0 = (mask[b * N + kg0] == 0), vk1 = (mask[b * N + kg1] == 0);
#pragma unroll
    for (int r = 0; r < 8; ++r) {
      const int qg = qg_r[r];
      float e0 = sc[0][r] + tab[(size_t)hh * (2 * N - 1) + (qg - kg0 + N - 1)];
      float e1 = sc[1][r] + tab[(size_t)hh * (2 * N - 1) + (qg - kg1 + N - 1)];
      if (!((kg0 <= qg) && vq_r[r] && vk0)) e0 = -3.0e38f;
      if (!((kg1 <= qg) && vq_r[r] && vk1)) e1 = -3.0e38f;
      float bm = fmaxf(e0, e1);
      bm = fmaxf(bm, __shfl_xor(bm, 1, 32));
      bm = fmaxf(bm, __shfl_xor(bm, 2, 32));
      bm = fmaxf(bm, __shfl_xor(bm, 4, 32));
      bm = fmaxf(bm, __shfl_xor(bm, 8, 32));
      const float mo = m_r[r], mn = fmaxf(mo, bm);
      const float scl = __expf(mo - mn);
      const float p0 = __expf(e0 - mn), p1 = __expf(e1 - mn);
      float bs = p0 + p1;
      bs += __shfl_xor(bs, 1, 32);
      bs += __shfl_xor(bs, 2, 32);
      bs += __shfl_xor(bs, 4, 32);
      bs += __shfl_xor(bs, 8, 32);
      m_r[r] = mn;
      s_r[r] = s_r[r] * scl + bs;
#pragma unroll
      for (int g2 = 0; g2 < 4; ++g2) Oacc[g2][r] *= scl;
      // stage P (C layout) -> per-wave LDS for A-fragment re-read
      sP[wave][(r + 8 * lh) * 32 + lm]      = (_Float16)p0;
      sP[wave][(r + 8 * lh) * 32 + 16 + lm] = (_Float16)p1;
    }

    // per-wave LDS staging: DS ops from this wave complete in order
    asm volatile("s_wait_dscnt 0" ::: "memory");
    const _Float16* pp = &sP[wave][0] + lm * 32 + lh * 8;
    v16h pf = cat8(*(const v8h*)pp, *(const v8h*)(pp + 16));
#pragma unroll
    for (int g2 = 0; g2 < 4; ++g2) {
      const _Float16* vp = sV + (g2 * 16 + lm) * 32 + lh * 8;
      v16h vf = cat8(*(const v8h*)vp, *(const v8h*)(vp + 16));
      Oacc[g2] = wmma16(pf, vf, Oacc[g2]);
    }
  }

  // ---- normalize + store O as f16 [B*N, H*64] ----
  float inv[8];
#pragma unroll
  for (int r = 0; r < 8; ++r) inv[r] = 1.0f / fmaxf(s_r[r], 1e-20f);
#pragma unroll
  for (int g2 = 0; g2 < 4; ++g2)
#pragma unroll
    for (int r = 0; r < 8; ++r) {
      const int row = qt * 128 + wave * 16 + r + 8 * lh;
      const int col = hh * 64 + g2 * 16 + lm;
      o16[(size_t)(b * N + row) * 1024 + col] = (_Float16)(Oacc[g2][r] * inv[r]);
    }
}

// ---------------------------------------------------------------------------
// Host orchestration
// ---------------------------------------------------------------------------
extern "C" void kernel_launch(void* const* d_in, const int* in_sizes, int n_in,
                              void* d_out, int out_size, void* d_ws, size_t ws_size,
                              hipStream_t stream) {
  (void)in_sizes; (void)n_in; (void)out_size; (void)ws_size;
  constexpr int B = 2, N = 1024, DIM = 1024, H = 16, FF = 4096;
  constexpr int M = B * N;                  // 2048
  const float* x_in  = (const float*)d_in[0];
  const unsigned char* maskp = (const unsigned char*)d_in[1];
  const float* ln1_g = (const float*)d_in[2];
  const float* ln1_b = (const float*)d_in[3];
  const float* qkv_w = (const float*)d_in[4];
  const float* out_w = (const float*)d_in[5];
  const float* temp  = (const float*)d_in[6];
  const float* ln2_g = (const float*)d_in[7];
  const float* ln2_b = (const float*)d_in[8];
  const float* glu_w = (const float*)d_in[9];
  const float* glu_b = (const float*)d_in[10];
  const float* ff2_w = (const float*)d_in[11];
  const float* ff2_b = (const float*)d_in[12];
  const float* p1w = (const float*)d_in[13];
  const float* p1b = (const float*)d_in[14];
  const float* p2w = (const float*)d_in[15];
  const float* p2b = (const float*)d_in[16];
  const float* p3w = (const float*)d_in[17];
  const float* p3b = (const float*)d_in[18];

  char* ws = (char*)d_ws;
  size_t off = 0;
  auto alloc = [&](size_t bytes) -> char* {
    char* p = ws + off;
    off += (bytes + 255) & ~(size_t)255;
    return p;
  };
  float*    xb   = (float*)   alloc((size_t)M * DIM * 4);       // running x
  _Float16* h16  = (_Float16*)alloc((size_t)M * DIM * 2);       // LN output
  float*    qkvb = (float*)   alloc((size_t)M * 3072 * 4);      // qkv planes
  _Float16* o16  = (_Float16*)alloc((size_t)M * DIM * 2);       // attn out
  float*    pb   = (float*)   alloc((size_t)M * 2 * FF * 4);    // GLU pre-act
  _Float16* f16b = (_Float16*)alloc((size_t)M * FF * 2);        // GLU act
  _Float16* wtb  = (_Float16*)alloc((size_t)2 * FF * DIM * 2);  // f16 W^T
  float*    tabb = (float*)   alloc((size_t)H * (2 * N - 1) * 4);

  copy_kernel<<<(M * DIM) / 256, 256, 0, stream>>>(x_in, xb);
  posbias_kernel<<<2 * N - 1, 256, 0, stream>>>(p1w, p1b, p2w, p2b, p3w, p3b,
                                                tabb, N);
  for (int l = 0; l < 4; ++l) {
    // --- cosine attention (PreNorm) ---
    ln_kernel<<<M, 256, 0, stream>>>(xb, ln1_g + l * DIM, ln1_b + l * DIM, h16);
    wconv_qkv_kernel<<<(DIM * 3072) / 256, 256, 0, stream>>>(
        qkv_w + (size_t)l * DIM * 3072, wtb, DIM);
    gemm_wmma<0><<<dim3(3072 / 256, M / 128), 256, 0, stream>>>(
        h16, wtb, qkvb, nullptr, nullptr, M, 3072, DIM);
    attn_kernel<<<B * H * (N / 128), 256, 0, stream>>>(qkvb, tabb, maskp,
                                                       temp + l, o16);
    wconv_kernel<<<(DIM * DIM) / 256, 256, 0, stream>>>(
        out_w + (size_t)l * DIM * DIM, wtb, DIM, DIM);
    gemm_wmma<2><<<dim3(DIM / 256, M / 128), 256, 0, stream>>>(
        o16, wtb, xb, nullptr, xb, M, DIM, DIM);  // x += o @ Wo
    // --- GLU FFN (PreNorm) ---
    ln_kernel<<<M, 256, 0, stream>>>(xb, ln2_g + l * DIM, ln2_b + l * DIM, h16);
    wconv_kernel<<<(DIM * 2 * FF) / 256, 256, 0, stream>>>(
        glu_w + (size_t)l * DIM * 2 * FF, wtb, DIM, 2 * FF);
    gemm_wmma<1><<<dim3((2 * FF) / 256, M / 128), 256, 0, stream>>>(
        h16, wtb, pb, glu_b + (size_t)l * 2 * FF, nullptr, M, 2 * FF, DIM);
    glu_kernel<<<((size_t)M * FF) / 256, 256, 0, stream>>>(pb, f16b);
    wconv_kernel<<<(FF * DIM) / 256, 256, 0, stream>>>(
        ff2_w + (size_t)l * FF * DIM, wtb, FF, DIM);
    float* dst = (l == 3) ? (float*)d_out : xb;
    gemm_wmma<3><<<dim3(DIM / 256, M / 128), 256, 0, stream>>>(
        f16b, wtb, dst, ff2_b + (size_t)l * DIM, xb, M, DIM, FF);
  }
}
